// GAT_65884798320867
// MI455X (gfx1250) — matrix operand order repaired and checked by reading, not compiled, not run on hardware
//
#include <hip/hip_runtime.h>

#define NNODES 4096
#define FIN    512
#define NHEADS 8
#define HID    64
#define HF     512   // NHEADS*HID
#define NCLS   64
#define LALPHA 0.2f

typedef __attribute__((ext_vector_type(16))) __bf16 v16bf;
typedef __attribute__((ext_vector_type(8)))  float  v8f;

union BfFrag { v16bf v; unsigned short u[16]; };

__device__ __forceinline__ unsigned short f2bf(float f) {
  union { float f; unsigned u; } cv; cv.f = f;
  unsigned u = cv.u;
  u += 0x7fffu + ((u >> 16) & 1u);   // round-to-nearest-even
  return (unsigned short)(u >> 16);
}

// Flat->LDS address: aperture maps LDS flat addresses by truncation to [31:0].
__device__ __forceinline__ unsigned lds_addr_of(const void* p) {
  return (unsigned)(unsigned long long)p;
}
// ASYNCcnt-tracked global->LDS copies (no VGPR round trip).
__device__ __forceinline__ void async_load_b64(unsigned lds_addr, const void* gaddr) {
  asm volatile("global_load_async_to_lds_b64 %0, %1, off"
               :: "v"(lds_addr), "v"(gaddr) : "memory");
}
__device__ __forceinline__ void async_load_b32(unsigned lds_addr, const void* gaddr) {
  asm volatile("global_load_async_to_lds_b32 %0, %1, off"
               :: "v"(lds_addr), "v"(gaddr) : "memory");
}
__device__ __forceinline__ void wait_async0() {
  asm volatile("s_wait_asynccnt 0x0" ::: "memory");
}

// ---------------------------------------------------------------- convert
__global__ void gat_cvt_bf16(const float* __restrict__ src,
                             unsigned short* __restrict__ dst, int n) {
  int t = blockIdx.x * blockDim.x + threadIdx.x;
  if (t < n) dst[t] = f2bf(src[t]);
}

// ---------------------------------------------------------------- GEMM C = A * B^T
// A: [M][K] bf16 row-major, B: [N][K] bf16 row-major (row n = output column n)
// C:  [M][ldc] f32 row-major.  CbT: [N][NNODES] bf16 (feature-major / transposed).
// One wave computes one 16x16 C tile. grid = (M/16, N/16), block = 32.
__global__ void gat_gemm_wmma(const unsigned short* __restrict__ A,
                              const unsigned short* __restrict__ B,
                              float* __restrict__ C,
                              unsigned short* __restrict__ CbT,
                              int K, int ldc) {
  const int lane = threadIdx.x & 31;
  const int m    = lane & 15;
  const int hi   = lane >> 4;
  const int rbase = blockIdx.x * 16;
  const int nbase = blockIdx.y * 16;
  v8f c = {};
  const unsigned short* arow = A + (size_t)(rbase + m) * K;
  const unsigned short* brow = B + (size_t)(nbase + m) * K;
  for (int k = 0; k < K; k += 32) {
    BfFrag af, bf;
    // A 16x32 bf16 layout: half t: K = t + (lane>=16 ? 16 : 0)
    const unsigned short* ap = arow + k + hi * 16;
#pragma unroll
    for (int t = 0; t < 8; ++t) { af.u[t] = ap[t]; af.u[t + 8] = ap[t + 8]; }
    // B 32x16 bf16 layout: N = lane%16, half t: K = t + (lane>=16 ? 16 : 0)
    const unsigned short* bp = brow + k + hi * 16;
#pragma unroll
    for (int t = 0; t < 16; ++t) bf.u[t] = bp[t];
    c = __builtin_amdgcn_wmma_f32_16x16x32_bf16(false, af.v, false, bf.v,
                                                (short)0, c, false, false);
  }
#pragma unroll
  for (int r = 0; r < 8; ++r) {
    int row = rbase + r + hi * 8;
    int col = nbase + m;
    C[(size_t)row * ldc + col] = c[r];
    if (CbT) CbT[(size_t)col * NNODES + row] = f2bf(c[r]);   // transposed bf16 copy
  }
}

// ---------------------------------------------------------------- wh1/wh2 dots
__global__ void gat_dot(const float* __restrict__ whC, int ldc,
                        const float* __restrict__ av1, const float* __restrict__ av2,
                        float* __restrict__ o1, float* __restrict__ o2v, int H) {
  int t = blockIdx.x * blockDim.x + threadIdx.x;
  if (t >= NNODES * H) return;
  int i = t / H, h = t - i * H;
  const float* row = whC + (size_t)i * ldc + h * 64;
  float s1 = 0.f, s2 = 0.f;
#pragma unroll 8
  for (int f = 0; f < 64; ++f) {
    s1 += row[f] * av1[h * 64 + f];
    s2 += row[f] * av2[h * 64 + f];
  }
  o1[h * NNODES + i] = s1;
  o2v[h * NNODES + i] = s2;
}

// ---------------------------------------------------------------- row max pass
// leaky is monotone: rowmax_e = leaky(wh1[i] + max_{j in adj_i} wh2[j])
__global__ void gat_rowmax(const int* __restrict__ adj,
                           const float* __restrict__ wh1a,
                           const float* __restrict__ wh2a,
                           float* __restrict__ mrow, int H) {
  __shared__ float red[8][256];
  const int i = blockIdx.x, tid = threadIdx.x;
  float pm[8];
#pragma unroll
  for (int h = 0; h < 8; ++h) pm[h] = -3e38f;
  for (int j = tid; j < NNODES; j += 256) {
    if (adj[(size_t)i * NNODES + j] > 0) {
      for (int h = 0; h < H; ++h) pm[h] = fmaxf(pm[h], wh2a[h * NNODES + j]);
    }
  }
  for (int h = 0; h < H; ++h) red[h][tid] = pm[h];
  __syncthreads();
  for (int s = 128; s > 0; s >>= 1) {
    if (tid < s)
      for (int h = 0; h < H; ++h)
        red[h][tid] = fmaxf(red[h][tid], red[h][tid + s]);
    __syncthreads();
  }
  if (tid < H) {
    float e = wh1a[tid * NNODES + i] + red[tid][0];
    mrow[tid * NNODES + i] = e > 0.f ? e : LALPHA * e;
  }
}

// ---------------------------------------------------------------- layer-1 flash aggregation
// block = 256 (8 waves, one head each), grid = NNODES/16 row blocks.
// adj tile + wh2 chunk staged via async global->LDS; B frags read straight
// from the transposed bf16 feature buffer (contiguous 32B per lane, L2-resident).
__global__ void gat_flash1(const int* __restrict__ adj,
                           const unsigned short* __restrict__ whbT,  // [HF][N] bf16
                           const float* __restrict__ wh1,            // [8][N]
                           const float* __restrict__ wh2,            // [8][N]
                           const float* __restrict__ mrow,           // [8][N]
                           float* __restrict__ hfeat,                // [N][HF]
                           unsigned short* __restrict__ hb) {        // [N][HF] bf16
  __shared__ int   s_adj[16 * 32];
  __shared__ float s_wh2[NHEADS * 32];
  __shared__ float s_srow[NHEADS * 16];

  const int tid = threadIdx.x;
  const int lane = tid & 31;
  const int h = tid >> 5;                 // wave id == head
  const int m = lane & 15;
  const int hi = lane >> 4;
  const int rbase = blockIdx.x * 16;

  const float wh1v = wh1[h * NNODES + rbase + m];
  const float mv   = mrow[h * NNODES + rbase + m];

  const int j2 = tid * 2;                 // this thread's 2 adj elements
  const unsigned adj_lds = lds_addr_of(&s_adj[j2]);
  const unsigned wh2_lds = lds_addr_of(&s_wh2[tid]);
  const size_t adj_row_off = (size_t)(rbase + (j2 >> 5)) * NNODES + (j2 & 31);
  const size_t wh2_off = (size_t)(tid >> 5) * NNODES + (tid & 31);

  v8f cacc[4] = {};
  float ssum = 0.f;

  for (int col0 = 0; col0 < NNODES; col0 += 32) {
    // async stage: 2 adj ints (b64) + 1 wh2 float (b32) per thread
    async_load_b64(adj_lds, adj + adj_row_off + col0);
    async_load_b32(wh2_lds, wh2 + wh2_off + col0);
    wait_async0();
    __syncthreads();

    // build p tile in A-fragment layout (this lane's row is m, halves are columns)
    BfFrag af;
#pragma unroll
    for (int t = 0; t < 16; ++t) {
      int cc = t + hi * 16;
      float p = 0.f;
      if (s_adj[m * 32 + cc] > 0) {
        float e = wh1v + s_wh2[h * 32 + cc];
        e = e > 0.f ? e : LALPHA * e;
        p = __expf(e - mv);
      }
      ssum += p;
      af.u[t] = f2bf(p);
    }
    // 4 feature tiles of 16: B frag = contiguous 16 bf16 from whbT (node-contiguous)
#pragma unroll
    for (int n = 0; n < 4; ++n) {
      BfFrag bf;
      const unsigned short* bp =
          whbT + (size_t)(h * HID + n * 16 + m) * NNODES + col0 + hi * 16;
#pragma unroll
      for (int t = 0; t < 16; ++t) bf.u[t] = bp[t];
      cacc[n] = __builtin_amdgcn_wmma_f32_16x16x32_bf16(false, af.v, false, bf.v,
                                                        (short)0, cacc[n], false, false);
    }
    __syncthreads();
  }

  ssum += __shfl_xor(ssum, 16);          // combine the two half-wave column groups
  if (lane < 16) s_srow[h * 16 + lane] = ssum;
  __syncthreads();

#pragma unroll
  for (int n = 0; n < 4; ++n)
#pragma unroll
    for (int r = 0; r < 8; ++r) {
      int rl = r + hi * 8;
      float v = cacc[n][r] / fmaxf(s_srow[h * 16 + rl], 1e-20f);
      v = v > 0.f ? v : __expf(v) - 1.f;   // ELU
      size_t idx = (size_t)(rbase + rl) * HF + h * HID + n * 16 + m;
      hfeat[idx] = v;
      hb[idx] = f2bf(v);
    }
}

// ---------------------------------------------------------------- layer-2 flash aggregation
// block = 256 (8 waves stripe the 4096 columns), grid = NNODES/16.
__global__ void gat_flash2(const int* __restrict__ adj,
                           const unsigned short* __restrict__ whb2T, // [NCLS][N] bf16
                           const float* __restrict__ wh1b,           // [N]
                           const float* __restrict__ wh2b,           // [N]
                           const float* __restrict__ mrow2,          // [N]
                           float* __restrict__ o2) {                 // [N][NCLS]
  __shared__ float redc[8][32][32];
  __shared__ float reds[8][32];
  __shared__ float stot[16];

  const int tid = threadIdx.x;
  const int lane = tid & 31;
  const int w = tid >> 5;
  const int m = lane & 15;
  const int hi = lane >> 4;
  const int rbase = blockIdx.x * 16;

  const float wh1v = wh1b[rbase + m];
  const float mv   = mrow2[rbase + m];
  const int* adjrow = adj + (size_t)(rbase + m) * NNODES;

  v8f cacc[4] = {};
  float ssum = 0.f;

  for (int ch = 0; ch < 16; ++ch) {
    const int col0 = w * 512 + ch * 32;
    if (ch + 1 < 16)
      __builtin_prefetch(adjrow + col0 + 32, 0, 1);   // global_prefetch_b8
    BfFrag af;
#pragma unroll
    for (int t = 0; t < 16; ++t) {
      int cc = t + hi * 16;
      int col = col0 + cc;
      float p = 0.f;
      if (adjrow[col] > 0) {
        float e = wh1v + wh2b[col];
        e = e > 0.f ? e : LALPHA * e;
        p = __expf(e - mv);
      }
      ssum += p;
      af.u[t] = f2bf(p);
    }
#pragma unroll
    for (int n = 0; n < 4; ++n) {
      BfFrag bf;
      const unsigned short* bp =
          whb2T + (size_t)(n * 16 + m) * NNODES + col0 + hi * 16;
#pragma unroll
      for (int t = 0; t < 16; ++t) bf.u[t] = bp[t];
      cacc[n] = __builtin_amdgcn_wmma_f32_16x16x32_bf16(false, af.v, false, bf.v,
                                                        (short)0, cacc[n], false, false);
    }
  }

  ssum += __shfl_xor(ssum, 16);
#pragma unroll
  for (int n = 0; n < 4; ++n)
#pragma unroll
    for (int r = 0; r < 8; ++r) redc[w][lane][n * 8 + r] = cacc[n][r];
  reds[w][lane] = ssum;
  __syncthreads();

  if (tid < 16) {
    float s = 0.f;
    for (int ww = 0; ww < 8; ++ww) s += reds[ww][tid];
    stot[tid] = fmaxf(s, 1e-20f);
  }
  __syncthreads();

#pragma unroll
  for (int q = 0; q < 4; ++q) {
    int pid = tid * 4 + q;                 // 0..1023 covers (lane2, slot)
    int l2 = pid >> 5, slot = pid & 31;
    int n = slot >> 3, r = slot & 7;
    float v = 0.f;
    for (int ww = 0; ww < 8; ++ww) v += redc[ww][l2][slot];
    int rl = r + ((l2 >> 4) << 3);
    v /= stot[rl];
    v = v > 0.f ? v : __expf(v) - 1.f;     // ELU
    o2[(size_t)(rbase + rl) * NCLS + n * 16 + (l2 & 15)] = v;
  }
}

// ---------------------------------------------------------------- log_softmax rows of 64
__global__ void gat_logsoftmax(const float* __restrict__ o2, float* __restrict__ out) {
  __shared__ float sm[64];
  const int row = blockIdx.x, t = threadIdx.x;
  float v = o2[(size_t)row * NCLS + t];
  sm[t] = v; __syncthreads();
  for (int s = 32; s > 0; s >>= 1) { if (t < s) sm[t] = fmaxf(sm[t], sm[t + s]); __syncthreads(); }
  float mx = sm[0]; __syncthreads();
  sm[t] = __expf(v - mx); __syncthreads();
  for (int s = 32; s > 0; s >>= 1) { if (t < s) sm[t] += sm[t + s]; __syncthreads(); }
  float lse = logf(sm[0]);
  out[(size_t)row * NCLS + t] = v - mx - lse;
}

// ---------------------------------------------------------------- launch
extern "C" void kernel_launch(void* const* d_in, const int* in_sizes, int n_in,
                              void* d_out, int out_size, void* d_ws, size_t ws_size,
                              hipStream_t stream) {
  const float* x   = (const float*)d_in[0];
  const int*   adj = (const int*)d_in[1];
  const float* W   = (const float*)d_in[2];
  const float* a1  = (const float*)d_in[3];
  const float* a2  = (const float*)d_in[4];
  const float* W2  = (const float*)d_in[5];
  const float* a21 = (const float*)d_in[6];
  const float* a22 = (const float*)d_in[7];
  float* out = (float*)d_out;

  char* ws = (char*)d_ws;
  size_t off = 0;
  auto alloc = [&](size_t bytes) -> void* {
    void* p = ws + off;
    off += (bytes + 255) & ~(size_t)255;
    return p;
  };
  unsigned short* xb    = (unsigned short*)alloc((size_t)NNODES * FIN * 2);
  unsigned short* Wb    = (unsigned short*)alloc((size_t)HF * FIN * 2);
  unsigned short* W2b   = (unsigned short*)alloc((size_t)NCLS * HF * 2);
  float*          wh    = (float*)alloc((size_t)NNODES * HF * 4);
  unsigned short* whbT  = (unsigned short*)alloc((size_t)HF * NNODES * 2);   // transposed
  float*          wh1   = (float*)alloc((size_t)NHEADS * NNODES * 4);
  float*          wh2   = (float*)alloc((size_t)NHEADS * NNODES * 4);
  float*          mrow  = (float*)alloc((size_t)NHEADS * NNODES * 4);
  float*          hfeat = (float*)alloc((size_t)NNODES * HF * 4);
  unsigned short* hb    = (unsigned short*)alloc((size_t)NNODES * HF * 2);
  float*          whc2  = (float*)alloc((size_t)NNODES * NCLS * 4);
  unsigned short* whb2T = (unsigned short*)alloc((size_t)NCLS * NNODES * 2); // transposed
  float*          wh1b  = (float*)alloc((size_t)NNODES * 4);
  float*          wh2b  = (float*)alloc((size_t)NNODES * 4);
  float*          mrow2 = (float*)alloc((size_t)NNODES * 4);
  float*          o2    = (float*)alloc((size_t)NNODES * NCLS * 4);

  // bf16 conversions
  gat_cvt_bf16<<<(NNODES * FIN + 255) / 256, 256, 0, stream>>>(x, xb, NNODES * FIN);
  gat_cvt_bf16<<<(HF * FIN + 255) / 256, 256, 0, stream>>>(W, Wb, HF * FIN);
  gat_cvt_bf16<<<(NCLS * HF + 255) / 256, 256, 0, stream>>>(W2, W2b, NCLS * HF);

  // layer 1
  gat_gemm_wmma<<<dim3(NNODES / 16, HF / 16), 32, 0, stream>>>(xb, Wb, wh, whbT, FIN, HF);
  gat_dot<<<(NNODES * NHEADS + 255) / 256, 256, 0, stream>>>(wh, HF, a1, a2, wh1, wh2, NHEADS);
  gat_rowmax<<<NNODES, 256, 0, stream>>>(adj, wh1, wh2, mrow, NHEADS);
  gat_flash1<<<NNODES / 16, 256, 0, stream>>>(adj, whbT, wh1, wh2, mrow, hfeat, hb);

  // layer 2
  gat_gemm_wmma<<<dim3(NNODES / 16, NCLS / 16), 32, 0, stream>>>(hb, W2b, whc2, whb2T, HF, NCLS);
  gat_dot<<<(NNODES + 255) / 256, 256, 0, stream>>>(whc2, NCLS, a21, a22, wh1b, wh2b, 1);
  gat_rowmax<<<NNODES, 256, 0, stream>>>(adj, wh1b, wh2b, mrow2, 1);
  gat_flash2<<<NNODES / 16, 256, 0, stream>>>(adj, whb2T, wh1b, wh2b, mrow2, o2);

  gat_logsoftmax<<<NNODES, 64, 0, stream>>>(o2, out);

  (void)in_sizes; (void)n_in; (void)out_size; (void)ws_size;
}
